// QSGDLinear_15058155339896
// MI455X (gfx1250) — matrix operand order repaired and checked by reading, not compiled, not run on hardware
//
#include <hip/hip_runtime.h>

// Problem constants (match reference)
constexpr int IN_F  = 4096;
constexpr int OUT_F = 16384;
constexpr int GS    = 128;

// Tiling: 128(M) x 256(N) block tile, 8 waves as 2(M) x 4(N), 64x64 per wave
constexpr int BM = 128;
constexpr int BN = 256;
constexpr int BK = 32;           // one f16 WMMA K step
constexpr int LDT = BK + 8;      // LDS row stride in halfs (80B -> conflict-free)

typedef _Float16 v16h __attribute__((ext_vector_type(16)));
typedef _Float16 v8h  __attribute__((ext_vector_type(8)));
typedef float    v8f  __attribute__((ext_vector_type(8)));
typedef float    v4f  __attribute__((ext_vector_type(4)));
typedef int      v4i  __attribute__((ext_vector_type(4)));

static __device__ __forceinline__ v16h combine16(v8h lo, v8h hi) {
    return __builtin_shufflevector(lo, hi, 0,1,2,3,4,5,6,7,8,9,10,11,12,13,14,15);
}

__global__ __launch_bounds__(256)
void ternary_linear_wmma(const float* __restrict__ x,
                         const int*   __restrict__ tern,
                         const float* __restrict__ scales,
                         float*       __restrict__ out) {
    __shared__ __align__(16) _Float16 sA[BM * LDT];  // x tile, f16, [m][k]
    __shared__ __align__(16) _Float16 sB[BN * LDT];  // W tile, f16, [n][k]

    const int t    = threadIdx.x;
    const int wave = t >> 5;
    const int lane = t & 31;
    const int l15  = lane & 15;
    const int lhi  = lane >> 4;          // 0 or 1 (half-wave)

    // blockIdx.x = M (fast, only 16) so concurrent m-blocks reuse the W slab in L2
    const int mblk = blockIdx.x * BM;
    const int nblk = blockIdx.y * BN;

    // ---- global staging mapping ----
    // A: 2 threads per row, 16 f32 each.  B: 1 thread per row, 32 int32 each.
    const int arow  = t >> 1;            // 0..127
    const int akoff = (t & 1) * 16;      // 0 or 16
    const float* xptr = x    + (size_t)(mblk + arow) * IN_F + akoff;
    const int*   wptr = tern + (size_t)(nblk + t) * IN_F;
    const float* sptr = scales + (size_t)(nblk + t) * (IN_F / GS);

    // ---- wave tile: 2 waves in M (64 rows), 4 waves in N (64 cols) ----
    const int wm = (wave >> 2) * 64;     // 0,64
    const int wn = (wave & 3) * 64;      // 0,64,128,192

    // LDS fragment base pointers (ISA layouts, CDNA5 ISA §7.12.2)
    // A 16x32 f16: lane l<16 -> row M=l, K chunks {0..7},{16..23}; l>=16 -> +8
    const _Float16* aB[4];
#pragma unroll
    for (int i = 0; i < 4; ++i)
        aB[i] = &sA[(wm + i * 16 + l15) * LDT + lhi * 8];
    // B 32x16 f16: lane l<16 -> col N=l, K chunks {0..7},{16..23}; l>=16 -> +8
    const _Float16* bB[4];
#pragma unroll
    for (int j = 0; j < 4; ++j)
        bB[j] = &sB[(wn + j * 16 + l15) * LDT + lhi * 8];

    v8f zero = {};
    v8f acc[4][4];
#pragma unroll
    for (int i = 0; i < 4; ++i)
#pragma unroll
        for (int j = 0; j < 4; ++j) acc[i][j] = zero;

    for (int kb = 0; kb < IN_F; kb += BK) {
        // ---- global loads (issue before barrier; overlap prior compute) ----
        v4f a0 = *(const v4f*)(xptr + kb + 0);
        v4f a1 = *(const v4f*)(xptr + kb + 4);
        v4f a2 = *(const v4f*)(xptr + kb + 8);
        v4f a3 = *(const v4f*)(xptr + kb + 12);
        v4i w0 = *(const v4i*)(wptr + kb + 0);
        v4i w1 = *(const v4i*)(wptr + kb + 4);
        v4i w2 = *(const v4i*)(wptr + kb + 8);
        v4i w3 = *(const v4i*)(wptr + kb + 12);
        v4i w4 = *(const v4i*)(wptr + kb + 16);
        v4i w5 = *(const v4i*)(wptr + kb + 20);
        v4i w6 = *(const v4i*)(wptr + kb + 24);
        v4i w7 = *(const v4i*)(wptr + kb + 28);
        const float s = sptr[kb >> 7];   // one scale per 128-wide K group

        __syncthreads();                 // prior iteration's LDS reads done

        // ---- convert + store A tile (f32 -> f16) ----
        v8h ha0, ha1;
#pragma unroll
        for (int j = 0; j < 4; ++j) {
            ha0[j]     = (_Float16)a0[j];
            ha0[j + 4] = (_Float16)a1[j];
            ha1[j]     = (_Float16)a2[j];
            ha1[j + 4] = (_Float16)a3[j];
        }
        *(v8h*)&sA[arow * LDT + akoff + 0] = ha0;
        *(v8h*)&sA[arow * LDT + akoff + 8] = ha1;

        // ---- dequant + store B tile (int ternary * scale -> f16) ----
        v8h hb0, hb1, hb2, hb3;
#pragma unroll
        for (int j = 0; j < 4; ++j) {
            hb0[j]     = (_Float16)((float)w0[j] * s);
            hb0[j + 4] = (_Float16)((float)w1[j] * s);
            hb1[j]     = (_Float16)((float)w2[j] * s);
            hb1[j + 4] = (_Float16)((float)w3[j] * s);
            hb2[j]     = (_Float16)((float)w4[j] * s);
            hb2[j + 4] = (_Float16)((float)w5[j] * s);
            hb3[j]     = (_Float16)((float)w6[j] * s);
            hb3[j + 4] = (_Float16)((float)w7[j] * s);
        }
        *(v8h*)&sB[t * LDT + 0]  = hb0;
        *(v8h*)&sB[t * LDT + 8]  = hb1;
        *(v8h*)&sB[t * LDT + 16] = hb2;
        *(v8h*)&sB[t * LDT + 24] = hb3;

        __syncthreads();                 // tile visible to all waves

        // ---- 16 WMMAs per wave: 64x64 of C, K += 32 ----
        v16h afrag[4];
#pragma unroll
        for (int i = 0; i < 4; ++i) {
            v8h lo = *(const v8h*)(aB[i]);
            v8h hi = *(const v8h*)(aB[i] + 16);
            afrag[i] = combine16(lo, hi);
        }
#pragma unroll
        for (int j = 0; j < 4; ++j) {
            v8h lo = *(const v8h*)(bB[j]);
            v8h hi = *(const v8h*)(bB[j] + 16);
            v16h bfrag = combine16(lo, hi);
#pragma unroll
            for (int i = 0; i < 4; ++i)
                acc[i][j] = __builtin_amdgcn_wmma_f32_16x16x32_f16(
                    false, afrag[i], false, bfrag, (short)0, acc[i][j], false, false);
        }
    }

    // ---- epilogue: C/D layout — VGPR r -> M = r (+8 upper half-wave), lane[3:0] -> N
    // Non-temporal stores: out is write-once; keep L2 for W/x reuse.
#pragma unroll
    for (int i = 0; i < 4; ++i) {
#pragma unroll
        for (int j = 0; j < 4; ++j) {
            const int m0 = mblk + wm + i * 16 + lhi * 8;
            const int n  = nblk + wn + j * 16 + l15;
#pragma unroll
            for (int r = 0; r < 8; ++r)
                __builtin_nontemporal_store(acc[i][j][r],
                                            &out[(size_t)(m0 + r) * OUT_F + n]);
        }
    }
}

extern "C" void kernel_launch(void* const* d_in, const int* in_sizes, int n_in,
                              void* d_out, int out_size, void* d_ws, size_t ws_size,
                              hipStream_t stream) {
    const float* x      = (const float*)d_in[0];
    const int*   tern   = (const int*)d_in[1];
    const float* scales = (const float*)d_in[2];
    float*       out    = (float*)d_out;

    const int M = in_sizes[0] / IN_F;    // 2048 (= 2*1024)
    dim3 grid(M / BM, OUT_F / BN);       // (16, 64): M fast -> W slab reuse in L2
    ternary_linear_wmma<<<grid, 256, 0, stream>>>(x, tern, scales, out);
}
